// NetworkMultiScale_21345987461181
// MI455X (gfx1250) — compile-verified, matrix operand order checked by reading, not run on hardware
//
#include <hip/hip_runtime.h>

// ---------------- problem constants ----------------
constexpr int Bc   = 2;
constexpr int Nc   = 20000;
constexpr int Sc   = 3000;
constexpr int Qc   = 5000;
constexpr int Lc   = 128;
constexpr int CINc = 3;
constexpr int KNN  = 16;
constexpr int NT   = Nc / 16;   // 1250 column tiles per batch

typedef __attribute__((ext_vector_type(16))) __bf16 v16bf;
typedef __attribute__((ext_vector_type(8)))  float  v8f;

// K-index inside a 32-wide block for 16-bit A/B WMMA fragments.
// element e (0..15), half = lane>>4:
//   pairs 0..3 -> K = 2*jp,       pairs 4..7 -> K = 16 + 2*(jp-4), +8*half +p
__device__ __forceinline__ int k_of(int e, int half) {
  int jp = e >> 1, p = e & 1;
  int base = (jp < 4) ? (2 * jp) : (16 + 2 * (jp - 4));
  return base + 8 * half + p;
}

// CDNA5 async global->LDS copy (ASYNCcnt-tracked).  lds_off = wave-relative
// LDS byte address (low 32 bits of the generic pointer), saddr = uniform
// 64-bit base, byte_off = per-lane signed byte offset (GVS mode).
__device__ __forceinline__ void async_load16(unsigned lds_off, const void* base,
                                             int byte_off) {
  asm volatile("global_load_async_to_lds_b128 %0, %1, %2"
               :: "v"(lds_off), "v"(byte_off), "s"(base)
               : "memory");
}

// ---------------- weight pre-swizzle: fp32 row-major -> bf16 fragment blobs --
// blob[((mt*KB + kb)*32 + lane)*16 + e] = (bf16) W[16*mt + lane%16][32*kb + k_of(e, lane/16)]
__global__ void prep_weights_kernel(const float* __restrict__ W,
                                    __bf16* __restrict__ blob,
                                    int OUT, int KIN) {
  int t = blockIdx.x * blockDim.x + threadIdx.x;
  int total = OUT * KIN;
  if (t >= total) return;
  int e    = t & 15;
  int lane = (t >> 4) & 31;
  int kbmt = t >> 9;
  int KB   = KIN >> 5;
  int kb   = kbmt % KB;
  int mt   = kbmt / KB;
  int row  = 16 * mt + (lane & 15);
  int col  = 32 * kb + k_of(e, lane >> 4);
  blob[t] = (__bf16)W[row * KIN + col];
}

// ---------------- backbone: lat = W2 @ relu(W1 @ x + b1) + b2 --------------
// One wave per 16-point column tile. Layer1 (K=3) per-lane VALU directly in
// B-fragment layout; layer2 (K=128) via 8x4 bf16 WMMAs.
__global__ __launch_bounds__(32) void backbone_kernel(
    const float* __restrict__ x, const float* __restrict__ W1,
    const float* __restrict__ b1, const __bf16* __restrict__ W2f,
    const float* __restrict__ b2, float* __restrict__ lat) {
  int tile = blockIdx.x % NT;
  int b    = blockIdx.x / NT;
  int lane = threadIdx.x;
  int col  = lane & 15;
  int half = lane >> 4;
  int n    = tile * 16 + col;

  const float* xb = x + (size_t)b * CINc * Nc;
  float x0 = xb[0 * Nc + n];
  float x1 = xb[1 * Nc + n];
  float x2 = xb[2 * Nc + n];

  // layer 1: build B fragments (4 K-blocks of 32) for h = relu(W1 x + b1)
  v16bf hfrag[4];
#pragma unroll
  for (int kb = 0; kb < 4; ++kb) {
#pragma unroll
    for (int e = 0; e < 16; ++e) {
      int c = 32 * kb + k_of(e, half);
      float h = b1[c] + W1[c * 3 + 0] * x0 + W1[c * 3 + 1] * x1 +
                W1[c * 3 + 2] * x2;
      hfrag[kb][e] = (__bf16)fmaxf(h, 0.0f);
    }
  }

  float* latb = lat + (size_t)b * Lc * Nc;
#pragma unroll
  for (int mt = 0; mt < 8; ++mt) {
    v8f acc = {};
#pragma unroll
    for (int kb = 0; kb < 4; ++kb) {
      v16bf a = *(const v16bf*)(W2f + (((mt * 4 + kb) * 32 + lane) << 4));
      acc = __builtin_amdgcn_wmma_f32_16x16x32_bf16(
          false, a, false, hfrag[kb], (short)0, acc, false, false);
    }
#pragma unroll
    for (int j = 0; j < 8; ++j) {
      int c = 16 * mt + 8 * half + j;
      latb[(size_t)c * Nc + n] = acc[j] + b2[c];
    }
  }
}

// ---------------- 1-NN upsample: gidx[b][n] = ids_down[argmin_s d2] --------
__global__ __launch_bounds__(256) void nn_upsample_kernel(
    const float* __restrict__ pos, const int* __restrict__ ids_down,
    int* __restrict__ gidx) {
  __shared__ float sx[1024], sy[1024], sz[1024];
  __shared__ int   sid[1024];
  int b = blockIdx.y;
  int n = blockIdx.x * 256 + threadIdx.x;
  const float* pb = pos + (size_t)b * 3 * Nc;

  float px = 0.f, py = 0.f, pz = 0.f;
  if (n < Nc) {
    px = pb[n]; py = pb[Nc + n]; pz = pb[2 * Nc + n];
  }
  float best = 3.4e38f;
  int bidx = 0;
  for (int base = 0; base < Sc; base += 1024) {
    int cnt = min(1024, Sc - base);
    if (base + 1024 < Sc)  // pull next index chunk toward the caches
      __builtin_prefetch(&ids_down[b * Sc + base + 1024 + threadIdx.x], 0, 0);
    for (int i = threadIdx.x; i < cnt; i += 256) {
      int s = ids_down[b * Sc + base + i];
      sid[i] = s;
      sx[i] = pb[s]; sy[i] = pb[Nc + s]; sz[i] = pb[2 * Nc + s];
    }
    __syncthreads();
    for (int i = 0; i < cnt; ++i) {
      float dx = px - sx[i], dy = py - sy[i], dz = pz - sz[i];
      float d2 = dx * dx + dy * dy + dz * dz;
      if (d2 < best) { best = d2; bidx = sid[i]; }
    }
    __syncthreads();
  }
  if (n < Nc) gidx[b * Nc + n] = bidx;
}

// ---------------- merge MLP: 256->128->128->128, all bf16 WMMA -------------
__global__ __launch_bounds__(32) void merge_kernel(
    const float* __restrict__ lat, const int* __restrict__ gidx,
    const __bf16* __restrict__ Wm1f, const float* __restrict__ bm1,
    const __bf16* __restrict__ Wm2f, const float* __restrict__ bm2,
    const __bf16* __restrict__ Wm3f, const float* __restrict__ bm3,
    float* __restrict__ merged) {
  __shared__ float hs[Lc * 16];
  int tile = blockIdx.x % NT;
  int b    = blockIdx.x / NT;
  int lane = threadIdx.x;
  int col  = lane & 15;
  int half = lane >> 4;
  int n    = tile * 16 + col;

  const float* latb = lat + (size_t)b * Lc * Nc;
  int g = gidx[b * Nc + n];

  // ---- layer 1: K=256 input = concat(lat[:,n], lat[:,g]) ----
  v16bf bfrag[8];
#pragma unroll
  for (int kb = 0; kb < 8; ++kb) {
#pragma unroll
    for (int e = 0; e < 16; ++e) {
      int kk = 32 * kb + k_of(e, half);
      float v = (kk < Lc) ? latb[(size_t)kk * Nc + n]
                          : latb[(size_t)(kk - Lc) * Nc + g];
      bfrag[kb][e] = (__bf16)v;
    }
  }
#pragma unroll
  for (int mt = 0; mt < 8; ++mt) {
    v8f acc = {};
#pragma unroll
    for (int kb = 0; kb < 8; ++kb) {
      v16bf a = *(const v16bf*)(Wm1f + (((mt * 8 + kb) * 32 + lane) << 4));
      acc = __builtin_amdgcn_wmma_f32_16x16x32_bf16(
          false, a, false, bfrag[kb], (short)0, acc, false, false);
    }
#pragma unroll
    for (int j = 0; j < 8; ++j) {
      int c = 16 * mt + 8 * half + j;
      hs[c * 16 + col] = fmaxf(acc[j] + bm1[c], 0.0f);
    }
  }
  __syncthreads();

  // ---- layer 2: K=128, read B-frags from LDS, keep C in regs ----
  v16bf bfrag2[4];
#pragma unroll
  for (int kb = 0; kb < 4; ++kb)
#pragma unroll
    for (int e = 0; e < 16; ++e)
      bfrag2[kb][e] = (__bf16)hs[(32 * kb + k_of(e, half)) * 16 + col];

  v8f acc2[8];
#pragma unroll
  for (int mt = 0; mt < 8; ++mt) {
    v8f acc = {};
#pragma unroll
    for (int kb = 0; kb < 4; ++kb) {
      v16bf a = *(const v16bf*)(Wm2f + (((mt * 4 + kb) * 32 + lane) << 4));
      acc = __builtin_amdgcn_wmma_f32_16x16x32_bf16(
          false, a, false, bfrag2[kb], (short)0, acc, false, false);
    }
    acc2[mt] = acc;
  }
  __syncthreads();
#pragma unroll
  for (int mt = 0; mt < 8; ++mt)
#pragma unroll
    for (int j = 0; j < 8; ++j) {
      int c = 16 * mt + 8 * half + j;
      hs[c * 16 + col] = fmaxf(acc2[mt][j] + bm2[c], 0.0f);
    }
  __syncthreads();

  // ---- layer 3: K=128, store to global (no relu) ----
  v16bf bfrag3[4];
#pragma unroll
  for (int kb = 0; kb < 4; ++kb)
#pragma unroll
    for (int e = 0; e < 16; ++e)
      bfrag3[kb][e] = (__bf16)hs[(32 * kb + k_of(e, half)) * 16 + col];

  float* mb = merged + (size_t)b * Lc * Nc;
#pragma unroll
  for (int mt = 0; mt < 8; ++mt) {
    v8f acc = {};
#pragma unroll
    for (int kb = 0; kb < 4; ++kb) {
      v16bf a = *(const v16bf*)(Wm3f + (((mt * 4 + kb) * 32 + lane) << 4));
      acc = __builtin_amdgcn_wmma_f32_16x16x32_bf16(
          false, a, false, bfrag3[kb], (short)0, acc, false, false);
    }
#pragma unroll
    for (int j = 0; j < 8; ++j) {
      int c = 16 * mt + 8 * half + j;
      mb[(size_t)c * Nc + n] = acc[j] + bm3[c];
    }
  }
}

// ---------------- projection: proj[b][o][n] = Wp @ merged + bp -------------
__global__ __launch_bounds__(256) void proj_kernel(
    const float* __restrict__ merged, const float* __restrict__ Wp,
    const float* __restrict__ bp, float* __restrict__ proj) {
  int t = blockIdx.x * blockDim.x + threadIdx.x;
  if (t >= Bc * Nc) return;
  int b = t / Nc;
  int n = t - b * Nc;
  const float* mb = merged + (size_t)b * Lc * Nc + n;
  float a0 = bp[0], a1 = bp[1];
#pragma unroll 8
  for (int c = 0; c < Lc; ++c) {
    float v = mb[(size_t)c * Nc];
    a0 += Wp[c] * v;
    a1 += Wp[Lc + c] * v;
  }
  proj[(size_t)b * 2 * Nc + n]      = a0;
  proj[(size_t)b * 2 * Nc + Nc + n] = a1;
}

// ---------------- decoder: top-16 NN + inverse-distance blend of proj ------
// Position tiles are double-buffered in LDS via CDNA5 async global->LDS
// copies: chunk c+1 streams in (ASYNCcnt) while the VALU scans chunk c.
constexpr int CH  = 1024;
constexpr int NCH = (Nc + CH - 1) / CH;   // 20 chunks (last = 544)

__global__ __launch_bounds__(256) void decoder_kernel(
    const float* __restrict__ pos, const float* __restrict__ posq,
    const float* __restrict__ proj, float* __restrict__ out) {
  __shared__ float sbuf[2][3][CH];
  int b = blockIdx.y;
  int q = blockIdx.x * 256 + threadIdx.x;
  const float* pb = pos + (size_t)b * 3 * Nc;
  const float* qb = posq + (size_t)b * 3 * Qc;

  float qx = 0.f, qy = 0.f, qz = 0.f;
  if (q < Qc) {
    qx = qb[q]; qy = qb[Qc + q]; qz = qb[2 * Qc + q];
  }
  float bd[KNN];
  int   bi[KNN];
#pragma unroll
  for (int k = 0; k < KNN; ++k) { bd[k] = 3.4e38f; bi[k] = 0; }

  // issue async loads of one chunk into buffer `buf` (3 x b128 per thread)
  auto issue = [&](int cidx, int buf) {
    int base = cidx * CH;
    int cnt  = min(CH, Nc - base);                 // always a multiple of 4
    int off4 = min(4 * (int)threadIdx.x, cnt - 4); // float offset, clamped
#pragma unroll
    for (int dim = 0; dim < 3; ++dim) {
      unsigned lds = (unsigned)(uintptr_t)&sbuf[buf][dim][off4];
      int gbyte = (dim * Nc + base + off4) * 4;
      async_load16(lds, pb, gbyte);
    }
  };

  issue(0, 0);
  int cur = 0;
  for (int c = 0; c < NCH; ++c) {
    bool more = (c + 1 < NCH);
    if (more) {
      issue(c + 1, cur ^ 1);
      // 3 new loads in flight; async loads complete in order, so <=3
      // guarantees the 3 loads for chunk c have landed in LDS.
      asm volatile("s_wait_asynccnt 0x3" ::: "memory");
    } else {
      asm volatile("s_wait_asynccnt 0x0" ::: "memory");
    }
    __syncthreads();

    int cnt = min(CH, Nc - c * CH);
    const float* bx = sbuf[cur][0];
    const float* by = sbuf[cur][1];
    const float* bz = sbuf[cur][2];
    for (int i = 0; i < cnt; ++i) {
      float dx = qx - bx[i], dy = qy - by[i], dz = qz - bz[i];
      float d2 = dx * dx + dy * dy + dz * dz;
      if (d2 < bd[KNN - 1]) {
        bd[KNN - 1] = d2;
        bi[KNN - 1] = c * CH + i;
#pragma unroll
        for (int p = KNN - 1; p > 0; --p) {
          if (bd[p] < bd[p - 1]) {
            float td = bd[p]; bd[p] = bd[p - 1]; bd[p - 1] = td;
            int   ti = bi[p]; bi[p] = bi[p - 1]; bi[p - 1] = ti;
          }
        }
      }
    }
    __syncthreads();   // readers done -> buffer reusable next iteration
    cur ^= 1;
  }

  if (q >= Qc) return;
  float w[KNN], wsum = 0.f;
#pragma unroll
  for (int k = 0; k < KNN; ++k) { w[k] = 1.0f / (bd[k] + 1e-8f); wsum += w[k]; }
  float inv = 1.0f / wsum;
  float o0 = 0.f, o1 = 0.f;
  const float* pj = proj + (size_t)b * 2 * Nc;
#pragma unroll
  for (int k = 0; k < KNN; ++k) {
    float wk = w[k] * inv;
    o0 += wk * pj[bi[k]];
    o1 += wk * pj[Nc + bi[k]];
  }
  out[(size_t)b * 2 * Qc + q]      = o0;
  out[(size_t)b * 2 * Qc + Qc + q] = o1;
}

// ---------------- host launcher --------------------------------------------
extern "C" void kernel_launch(void* const* d_in, const int* in_sizes, int n_in,
                              void* d_out, int out_size, void* d_ws,
                              size_t ws_size, hipStream_t stream) {
  const float* x        = (const float*)d_in[0];
  const float* pos      = (const float*)d_in[1];
  const float* posq     = (const float*)d_in[2];
  const int*   ids_down = (const int*)d_in[3];
  const float* W1  = (const float*)d_in[4];
  const float* b1  = (const float*)d_in[5];
  const float* W2  = (const float*)d_in[6];
  const float* b2  = (const float*)d_in[7];
  const float* Wm1 = (const float*)d_in[8];
  const float* bm1 = (const float*)d_in[9];
  const float* Wm2 = (const float*)d_in[10];
  const float* bm2 = (const float*)d_in[11];
  const float* Wm3 = (const float*)d_in[12];
  const float* bm3 = (const float*)d_in[13];
  const float* Wp  = (const float*)d_in[14];
  const float* bp  = (const float*)d_in[15];
  float* out = (float*)d_out;
  char*  ws  = (char*)d_ws;

  const size_t latBytes = (size_t)Bc * Lc * Nc * sizeof(float);   // 20.48 MB
  size_t off = 0;
  float* lat    = (float*)(ws + off); off += latBytes;
  float* merged = (float*)(ws + off); off += latBytes;
  int*   gidx   = (int*)(ws + off);   off += (size_t)Bc * Nc * sizeof(int);
  float* proj   = (float*)(ws + off); off += (size_t)Bc * 2 * Nc * sizeof(float);
  __bf16* w2f  = (__bf16*)(ws + off); off += (size_t)Lc * Lc * sizeof(__bf16);
  __bf16* wm1f = (__bf16*)(ws + off); off += (size_t)Lc * 2 * Lc * sizeof(__bf16);
  __bf16* wm2f = (__bf16*)(ws + off); off += (size_t)Lc * Lc * sizeof(__bf16);
  __bf16* wm3f = (__bf16*)(ws + off); off += (size_t)Lc * Lc * sizeof(__bf16);

  // 1) weight pre-swizzle into bf16 WMMA fragment order
  prep_weights_kernel<<<(Lc * Lc + 255) / 256, 256, 0, stream>>>(W2, w2f, Lc, Lc);
  prep_weights_kernel<<<(Lc * 2 * Lc + 255) / 256, 256, 0, stream>>>(Wm1, wm1f, Lc, 2 * Lc);
  prep_weights_kernel<<<(Lc * Lc + 255) / 256, 256, 0, stream>>>(Wm2, wm2f, Lc, Lc);
  prep_weights_kernel<<<(Lc * Lc + 255) / 256, 256, 0, stream>>>(Wm3, wm3f, Lc, Lc);

  // 2) full-resolution backbone (lat_down is a gather of lat -> computed once)
  backbone_kernel<<<Bc * NT, 32, 0, stream>>>(x, W1, b1, w2f, b2, lat);

  // 3) 1-NN upsample index (composed with ids_down)
  nn_upsample_kernel<<<dim3((Nc + 255) / 256, Bc), 256, 0, stream>>>(pos, ids_down, gidx);

  // 4) merge MLP (bf16 WMMA, 3 layers)
  merge_kernel<<<Bc * NT, 32, 0, stream>>>(lat, gidx, wm1f, bm1, wm2f, bm2,
                                           wm3f, bm3, merged);

  // 5) pre-projection 128 -> 2 (linear, so it commutes with the kNN blend)
  proj_kernel<<<(Bc * Nc + 255) / 256, 256, 0, stream>>>(merged, Wp, bp, proj);

  // 6) top-16 kNN inverse-distance interpolation on projected features
  decoder_kernel<<<dim3((Qc + 255) / 256, Bc), 256, 0, stream>>>(pos, posq, proj, out);
}
